// Transformer_71957882077842
// MI455X (gfx1250) — matrix operand order repaired and checked by reading, not compiled, not run on hardware
//
#include <hip/hip_runtime.h>
#include <hip/hip_bf16.h>

// ---------------- model dimensions (compile-time) ----------------
#define LAY   8
#define HD    1024      // hidden
#define KD    1024      // key dim
#define FF    4096      // mlp intermediate
#define VOC   32000
#define NB    2
#define SEQ   2048
#define MTOK  (NB*SEQ)  // 4096 token rows

typedef __attribute__((ext_vector_type(16))) __bf16        v16bf;
typedef __attribute__((ext_vector_type(8)))  float         v8f;
typedef __attribute__((ext_vector_type(8)))  unsigned int  v8u;
typedef __attribute__((ext_vector_type(4)))  unsigned int  v4u;
typedef __attribute__((ext_vector_type(4)))  float         v4f;

__device__ __forceinline__ unsigned short f2b(float f) {
    unsigned int u = __float_as_uint(f);
    unsigned int r = u + 0x7FFFu + ((u >> 16) & 1u);   // round-to-nearest-even
    return (unsigned short)(r >> 16);
}

// LDS tile row stride: 40 shorts = 80 bytes (16B aligned rows, 20 dwords,
// m*20 mod 64 distinct for m=0..15 -> conflict-free fragment gathers)
#define LSTR 40

// =====================================================================
// bf16 x bf16 -> fp32 tiled WMMA GEMM, double-buffered LDS.
//   TRANS_B=0 : B is [K,N] bf16 row-major;  TRANS_B=1 : B is [N,K] (A*B^T)
// Block = 256 threads (8 wave32), tile 128x128, K-step 32.
// Each wave: 64x32 sub-tile = 4x2 WMMA 16x16x32 bf16, fp32 accum.
// Epilogue: (+bias fp32) (+res fp32) (relu), writes fp32 C and/or bf16 Cb.
// =====================================================================
template<bool TRANS_B, bool RELU, bool HAS_BIAS, bool HAS_RES, bool OUT_F32, bool OUT_BF>
__global__ __launch_bounds__(256, 2)
void gemm_bf16_wmma(const unsigned short* __restrict__ Ag,
                    const unsigned short* __restrict__ Bg,
                    const float* __restrict__ bias, const float* __restrict__ Res,
                    float* __restrict__ C, unsigned short* __restrict__ Cb,
                    int Kdim, int lda, int ldb, int ldc,
                    long long sA, long long sB, long long sC)
{
    const int batch = blockIdx.z;
    Ag += (size_t)batch * sA;
    Bg += (size_t)batch * sB;
    if (OUT_F32) C   += (size_t)batch * sC;
    if (OUT_BF)  Cb  += (size_t)batch * sC;
    if (HAS_RES) Res += (size_t)batch * sC;

    const int bm = blockIdx.y * 128;
    const int bn = blockIdx.x * 128;

    __shared__ unsigned short As[2][128][LSTR];   // [m][k]
    __shared__ unsigned short Bs[2][128][LSTR];   // [n][k]

    const int tid    = threadIdx.x;
    const int lane   = tid & 31;
    const int wave   = tid >> 5;
    const int wm     = (wave & 1)  * 64;
    const int wn     = (wave >> 1) * 32;
    const int lane15 = lane & 15;
    const int kh     = lane >> 4;

    v8f acc[4][2];
    #pragma unroll
    for (int mi = 0; mi < 4; ++mi)
        #pragma unroll
        for (int ni = 0; ni < 2; ++ni)
            #pragma unroll
            for (int r = 0; r < 8; ++r) acc[mi][ni][r] = 0.0f;

    // ---- staging thread mapping ----
    // A: 128x32 bf16 = 512 uint4; thread handles f = tid, tid+256
    //    m = f>>2, q = f&3  -> global row bm+m, cols k0+8q..+7
    // B non-trans: thread owns ng = tid&15 (8 n's), kp = tid>>4 (k pair)
    // B trans:     same mapping as A (n rows contiguous in k)
    const int am0 = tid >> 2,        aq0 = tid & 3;
    const int am1 = (tid + 256) >> 2, aq1 = (tid + 256) & 3;
    const int bng = tid & 15,        bkp = tid >> 4;

    v4u ra[2], rb[2];

    auto load_tiles = [&](int k0) {
        ra[0] = *reinterpret_cast<const v4u*>(Ag + (size_t)(bm + am0) * lda + k0 + 8 * aq0);
        ra[1] = *reinterpret_cast<const v4u*>(Ag + (size_t)(bm + am1) * lda + k0 + 8 * aq1);
        if (TRANS_B) {
            rb[0] = *reinterpret_cast<const v4u*>(Bg + (size_t)(bn + am0) * ldb + k0 + 8 * aq0);
            rb[1] = *reinterpret_cast<const v4u*>(Bg + (size_t)(bn + am1) * ldb + k0 + 8 * aq1);
        } else {
            rb[0] = *reinterpret_cast<const v4u*>(Bg + (size_t)(k0 + 2 * bkp)     * ldb + bn + 8 * bng);
            rb[1] = *reinterpret_cast<const v4u*>(Bg + (size_t)(k0 + 2 * bkp + 1) * ldb + bn + 8 * bng);
        }
    };
    auto store_tiles = [&](int buf) {
        *reinterpret_cast<v4u*>(&As[buf][am0][8 * aq0]) = ra[0];
        *reinterpret_cast<v4u*>(&As[buf][am1][8 * aq1]) = ra[1];
        if (TRANS_B) {
            *reinterpret_cast<v4u*>(&Bs[buf][am0][8 * aq0]) = rb[0];
            *reinterpret_cast<v4u*>(&Bs[buf][am1][8 * aq1]) = rb[1];
        } else {
            unsigned int* BsU = reinterpret_cast<unsigned int*>(&Bs[buf][0][0]);
            #pragma unroll
            for (int j = 0; j < 4; ++j) {
                unsigned int d = rb[0][j], e = rb[1][j];
                unsigned int lo = (d & 0xFFFFu) | (e << 16);          // n=8ng+2j
                unsigned int hi = (d >> 16)     | (e & 0xFFFF0000u);  // n=8ng+2j+1
                BsU[(8 * bng + 2 * j)     * (LSTR / 2) + bkp] = lo;
                BsU[(8 * bng + 2 * j + 1) * (LSTR / 2) + bkp] = hi;
            }
        }
    };
    auto compute = [&](int buf) {
        const unsigned int* AsU = reinterpret_cast<const unsigned int*>(&As[buf][0][0]);
        const unsigned int* BsU = reinterpret_cast<const unsigned int*>(&Bs[buf][0][0]);
        v16bf afrag[4];
        #pragma unroll
        for (int mi = 0; mi < 4; ++mi) {
            const int arow = wm + mi * 16 + lane15;
            v8u t;
            #pragma unroll
            for (int v = 0; v < 8; ++v) {
                // ISA 16-bit A 16x32 layout: kpair = (v&3) + 4*kh + 8*(v>>2)
                int kp = (v & 3) + (kh << 2) + ((v >> 2) << 3);
                t[v] = AsU[arow * (LSTR / 2) + kp];
            }
            afrag[mi] = __builtin_bit_cast(v16bf, t);
        }
        v16bf bfrag[2];
        #pragma unroll
        for (int ni = 0; ni < 2; ++ni) {
            const int bcol = wn + ni * 16 + lane15;
            v8u t;
            #pragma unroll
            for (int v = 0; v < 8; ++v) {
                // B KxN layout (matches A): kpair = (v&3) + 4*kh + 8*(v>>2)
                int kp = (v & 3) + (kh << 2) + ((v >> 2) << 3);
                t[v] = BsU[bcol * (LSTR / 2) + kp];
            }
            bfrag[ni] = __builtin_bit_cast(v16bf, t);
        }
        #pragma unroll
        for (int mi = 0; mi < 4; ++mi)
            #pragma unroll
            for (int ni = 0; ni < 2; ++ni)
                acc[mi][ni] = __builtin_amdgcn_wmma_f32_16x16x32_bf16(
                    false, afrag[mi], false, bfrag[ni],
                    (short)0, acc[mi][ni], false, false);
    };

    // ---- pipelined main loop (single barrier per K-step) ----
    const int nk = Kdim >> 5;
    load_tiles(0);
    store_tiles(0);
    __syncthreads();
    for (int it = 0; it < nk; ++it) {
        if (it + 1 < nk) load_tiles((it + 1) << 5);
        if (it + 2 < nk) {
            __builtin_prefetch(Ag + (size_t)(bm + am0) * lda + ((it + 2) << 5), 0, 1);
            __builtin_prefetch(Bg + (TRANS_B ? (size_t)(bn + am0) * ldb + ((it + 2) << 5)
                                             : (size_t)(((it + 2) << 5) + 2 * bkp) * ldb + bn + 8 * bng), 0, 1);
        }
        compute(it & 1);
        if (it + 1 < nk) store_tiles((it + 1) & 1);
        __syncthreads();
    }

    // ---- epilogue ----
    const int r0 = kh * 8;
    #pragma unroll
    for (int mi = 0; mi < 4; ++mi) {
        #pragma unroll
        for (int ni = 0; ni < 2; ++ni) {
            const int col = bn + wn + ni * 16 + lane15;
            #pragma unroll
            for (int r = 0; r < 8; ++r) {
                const int row = bm + wm + mi * 16 + r0 + r;
                float v = acc[mi][ni][r];
                if (HAS_BIAS) v += bias[col];
                if (HAS_RES)  v += Res[(size_t)row * ldc + col];
                if (RELU)     v = fmaxf(v, 0.0f);
                if (OUT_F32)  C[(size_t)row * ldc + col]  = v;
                if (OUT_BF)   Cb[(size_t)row * ldc + col] = f2b(v);
            }
        }
    }
}

// =====================================================================
// fp32 -> bf16 bulk convert (count multiple of 1024; 4 elems/thread)
// =====================================================================
__global__ __launch_bounds__(256)
void cvt_f32_bf16(const float* __restrict__ src, unsigned short* __restrict__ dst)
{
    const size_t i = ((size_t)blockIdx.x * 256 + threadIdx.x) * 4;
    v4f v = *reinterpret_cast<const v4f*>(src + i);
    unsigned int lo = (unsigned int)f2b(v[0]) | ((unsigned int)f2b(v[1]) << 16);
    unsigned int hi = (unsigned int)f2b(v[2]) | ((unsigned int)f2b(v[3]) << 16);
    *reinterpret_cast<unsigned int*>(dst + i)     = lo;
    *reinterpret_cast<unsigned int*>(dst + i + 2) = hi;
}

// =====================================================================
// Causal softmax: scores fp32 [b][s][t] -> attn bf16, t<=s, scaled.
// =====================================================================
__global__ __launch_bounds__(256)
void softmax_causal(const float* __restrict__ scores, unsigned short* __restrict__ attn,
                    float scale)
{
    const int row = blockIdx.x;            // 0 .. NB*SEQ-1
    const int b   = row / SEQ;
    const int s   = row % SEQ;
    const float* p   = scores + (size_t)b * SEQ * SEQ + (size_t)s * SEQ;
    unsigned short* q = attn  + (size_t)b * SEQ * SEQ + (size_t)s * SEQ;
    const int len = s + 1;
    const int tid = threadIdx.x;
    __shared__ float red[256];

    float m = -1e30f;
    for (int t = tid; t < len; t += 256) m = fmaxf(m, p[t] * scale);
    red[tid] = m; __syncthreads();
    #pragma unroll
    for (int off = 128; off > 0; off >>= 1) {
        if (tid < off) red[tid] = fmaxf(red[tid], red[tid + off]);
        __syncthreads();
    }
    m = red[0]; __syncthreads();

    float sum = 0.0f;
    for (int t = tid; t < len; t += 256) sum += __expf(p[t] * scale - m);
    red[tid] = sum; __syncthreads();
    #pragma unroll
    for (int off = 128; off > 0; off >>= 1) {
        if (tid < off) red[tid] += red[tid + off];
        __syncthreads();
    }
    const float inv = 1.0f / red[0];

    for (int t = tid; t < len; t += 256) q[t] = f2b(__expf(p[t] * scale - m) * inv);
    for (int t = len + tid; t < SEQ; t += 256) q[t] = 0;
}

// =====================================================================
// Embedding gather: fp32 master x + bf16 copy
// =====================================================================
__global__ __launch_bounds__(256)
void embed_gather(const int* __restrict__ tokens, const float* __restrict__ emb,
                  float* __restrict__ x, unsigned short* __restrict__ xb)
{
    const int row = blockIdx.x;
    const int tok = tokens[row];
    const float* src = emb + (size_t)tok * HD;
    float* dst = x + (size_t)row * HD;
    unsigned short* dstb = xb + (size_t)row * HD;
    for (int h = threadIdx.x; h < HD; h += blockDim.x) {
        float v = src[h];
        dst[h]  = v;
        dstb[h] = f2b(v);
    }
}

// =====================================================================
extern "C" void kernel_launch(void* const* d_in, const int* in_sizes, int n_in,
                              void* d_out, int out_size, void* d_ws, size_t ws_size,
                              hipStream_t stream)
{
    const int*   tokens    = (const int*)  d_in[0];
    const float* embedding = (const float*)d_in[1];
    const float* qkv_w     = (const float*)d_in[2];   // [L,H,3K]
    const float* o_w       = (const float*)d_in[3];   // [L,K,H]
    const float* up_w      = (const float*)d_in[4];   // [L,H,I]
    const float* up_b      = (const float*)d_in[5];   // [L,I]
    const float* down_w    = (const float*)d_in[6];   // [L,I,H]
    const float* down_b    = (const float*)d_in[7];   // [L,H]
    const float* unemb_w   = (const float*)d_in[8];   // [H,V]
    const float* unemb_b   = (const float*)d_in[9];   // [V]
    float* logits = (float*)d_out;

    // ---- workspace carve-up ----
    char* w = (char*)d_ws;
    float* x         = (float*)w;          w += (size_t)MTOK * HD * 4;        // fp32 residual master
    float* scores    = (float*)w;          w += (size_t)NB * SEQ * SEQ * 4;   // fp32 attn scores
    unsigned short* x_bf    = (unsigned short*)w;  w += (size_t)MTOK * HD * 2;
    unsigned short* qkv_bf  = (unsigned short*)w;  w += (size_t)MTOK * 3 * KD * 2;
    unsigned short* attn_bf = (unsigned short*)w;  w += (size_t)NB * SEQ * SEQ * 2;
    unsigned short* o_bf    = (unsigned short*)w;  w += (size_t)MTOK * KD * 2;
    unsigned short* hmid_bf = (unsigned short*)w;  w += (size_t)MTOK * FF * 2;
    unsigned short* w_bf    = (unsigned short*)w;  w += (size_t)HD * VOC * 2; // weight scratch (max size)

    const float scale = 1.0f / 32.0f;   // 1/sqrt(1024)
    const dim3 blk(256);

    embed_gather<<<MTOK, blk, 0, stream>>>(tokens, embedding, x, x_bf);

    for (int l = 0; l < LAY; ++l) {
        const float* wqkv = qkv_w  + (size_t)l * HD * 3 * KD;
        const float* wo   = o_w    + (size_t)l * KD * HD;
        const float* wu   = up_w   + (size_t)l * HD * FF;
        const float* bu   = up_b   + (size_t)l * FF;
        const float* wd   = down_w + (size_t)l * FF * HD;
        const float* bd   = down_b + (size_t)l * HD;

        // qkv = x @ Wqkv -> bf16 [MTOK, 3K]
        cvt_f32_bf16<<<(HD * 3 * KD) / 1024, blk, 0, stream>>>(wqkv, w_bf);
        gemm_bf16_wmma<false,false,false,false,false,true>
            <<<dim3(3*KD/128, MTOK/128, 1), blk, 0, stream>>>(
            x_bf, w_bf, nullptr, nullptr, nullptr, qkv_bf,
            HD, HD, 3*KD, 3*KD, 0, 0, 0);

        // scores = q @ k^T (per batch, NT) -> fp32
        gemm_bf16_wmma<true,false,false,false,true,false>
            <<<dim3(SEQ/128, SEQ/128, NB), blk, 0, stream>>>(
            qkv_bf, qkv_bf + KD, nullptr, nullptr, scores, nullptr,
            KD, 3*KD, 3*KD, SEQ,
            (long long)SEQ*3*KD, (long long)SEQ*3*KD, (long long)SEQ*SEQ);

        // causal softmax (fp32 in, bf16 probs out)
        softmax_causal<<<NB*SEQ, blk, 0, stream>>>(scores, attn_bf, scale);

        // o = attn @ v (per batch) -> bf16
        gemm_bf16_wmma<false,false,false,false,false,true>
            <<<dim3(KD/128, SEQ/128, NB), blk, 0, stream>>>(
            attn_bf, qkv_bf + 2*KD, nullptr, nullptr, nullptr, o_bf,
            SEQ, SEQ, 3*KD, KD,
            (long long)SEQ*SEQ, (long long)SEQ*3*KD, (long long)SEQ*KD);

        // x = x + o @ Wo  (fp32 master + bf16 copy)
        cvt_f32_bf16<<<(KD * HD) / 1024, blk, 0, stream>>>(wo, w_bf);
        gemm_bf16_wmma<false,false,false,true,true,true>
            <<<dim3(HD/128, MTOK/128, 1), blk, 0, stream>>>(
            o_bf, w_bf, nullptr, x, x, x_bf,
            KD, KD, HD, HD, 0, 0, 0);

        // hmid = relu(x @ Wu + bu) -> bf16
        cvt_f32_bf16<<<(HD * FF) / 1024, blk, 0, stream>>>(wu, w_bf);
        gemm_bf16_wmma<false,true,true,false,false,true>
            <<<dim3(FF/128, MTOK/128, 1), blk, 0, stream>>>(
            x_bf, w_bf, bu, nullptr, nullptr, hmid_bf,
            HD, HD, FF, FF, 0, 0, 0);

        // x = x + hmid @ Wd + bd  (fp32 master + bf16 copy)
        cvt_f32_bf16<<<(FF * HD) / 1024, blk, 0, stream>>>(wd, w_bf);
        gemm_bf16_wmma<false,false,true,true,true,true>
            <<<dim3(HD/128, MTOK/128, 1), blk, 0, stream>>>(
            hmid_bf, w_bf, bd, x, x, x_bf,
            FF, FF, HD, HD, 0, 0, 0);
    }

    // logits = x @ Wun + bun -> fp32 [MTOK, VOC]
    cvt_f32_bf16<<<(HD * VOC) / 1024, blk, 0, stream>>>(unemb_w, w_bf);
    gemm_bf16_wmma<false,false,true,false,true,false>
        <<<dim3(VOC/128, MTOK/128, 1), blk, 0, stream>>>(
        x_bf, w_bf, unemb_b, nullptr, logits, nullptr,
        HD, HD, VOC, VOC, 0, 0, 0);

    (void)in_sizes; (void)n_in; (void)out_size; (void)ws_size;
}
